// MultiheadAttention_3255585210705
// MI455X (gfx1250) — matrix operand order repaired and checked by reading, not compiled, not run on hardware
//
#include <hip/hip_runtime.h>

typedef __attribute__((ext_vector_type(16))) __bf16 v16bf;
typedef __attribute__((ext_vector_type(8)))  float  v8f;
typedef unsigned short u16;
typedef unsigned int   u32;

static __device__ __forceinline__ u16 f32_to_bf16(float f) {
  u32 u = __float_as_uint(f);
  u32 r = u + 0x7FFFu + ((u >> 16) & 1u);   // round-to-nearest-even
  return (u16)(r >> 16);
}

union Frag {
  uint4 q[2];
  v16bf v;
};

static __device__ __forceinline__ v8f wmma_bf16(v16bf a, v16bf b, v8f c) {
  return __builtin_amdgcn_wmma_f32_16x16x32_bf16(false, a, false, b, (short)0, c,
                                                 false, false);
}

// gfx1250 async global->LDS copy (ASYNCcnt-tracked), 16 bytes per lane.
// GVS addressing: mem = SADDR(64b sgpr pair) + VADDR(32b byte offset).
static __device__ __forceinline__ void async_b128(u32 lds_addr, u32 voff,
                                                  const void* sbase) {
  asm volatile("global_load_async_to_lds_b128 %0, %1, %2"
               :: "v"(lds_addr), "v"(voff), "s"(sbase)
               : "memory");
}

// ---------------------------------------------------------------- convert
__global__ void cvt_f32_bf16(const float* __restrict__ src, u16* __restrict__ dst, int n) {
  int i = blockIdx.x * blockDim.x + threadIdx.x;
  int stride = gridDim.x * blockDim.x;
  for (; i < n; i += stride) dst[i] = f32_to_bf16(src[i]);
}

// ---------------------------------------------------------------- GEMM core
// C[M,N] = A[M,K] * Bt[N,K]^T, bf16 in, f32 accum.
// Block 128x128, 256 threads (8 waves), wave tile 64x32 (4x2 WMMA tiles).
// Double-buffered LDS filled by async global->LDS DMA, overlapped with WMMA.
#define GEMM_K 1024

#define GEMM_MAINLOOP(A_, Bt_)                                                   \
  __shared__ __align__(16) u16 At[2][128 * 32];                                  \
  __shared__ __align__(16) u16 Bl[2][128 * 32];                                  \
  const int t = threadIdx.x;                                                     \
  const int lane = t & 31, wave = t >> 5;                                        \
  const int mo = (wave & 1) * 64, no = (wave >> 1) * 32;                         \
  const int nBlk = blockIdx.x * 128, mBlk = blockIdx.y * 128;                    \
  const int ln = lane & 15, lh = lane >> 4;                                      \
  const int lr = t >> 1, lc = (t & 1) * 16;                                      \
  const u32 aLds0 = (u32)(size_t)(const void*)(&At[0][lr * 32 + lc]);            \
  const u32 bLds0 = (u32)(size_t)(const void*)(&Bl[0][lr * 32 + lc]);            \
  const u32 bufStride = (u32)(128 * 32 * 2);                                     \
  const u32 vaA = (u32)(((mBlk + lr) * GEMM_K + lc) * 2);                        \
  const u32 vaB = (u32)(((nBlk + lr) * GEMM_K + lc) * 2);                        \
  v8f acc[4][2] = {};                                                            \
  /* prologue: DMA tile 0 into buffer 0 */                                       \
  async_b128(aLds0,      vaA,      (const void*)(A_));                           \
  async_b128(aLds0 + 16, vaA + 16, (const void*)(A_));                           \
  async_b128(bLds0,      vaB,      (const void*)(Bt_));                          \
  async_b128(bLds0 + 16, vaB + 16, (const void*)(Bt_));                          \
  for (int kb = 0; kb < GEMM_K; kb += 32) {                                      \
    const int buf = (kb >> 5) & 1;                                               \
    __syncthreads(); /* all waves done reading buf^1 from prev iter */           \
    if (kb + 32 < GEMM_K) {                                                      \
      const u32 aL = aLds0 + (u32)(buf ^ 1) * bufStride;                         \
      const u32 bL = bLds0 + (u32)(buf ^ 1) * bufStride;                         \
      const u32 off = (u32)(kb + 32) * 2;                                        \
      async_b128(aL,      vaA + off,      (const void*)(A_));                    \
      async_b128(aL + 16, vaA + off + 16, (const void*)(A_));                    \
      async_b128(bL,      vaB + off,      (const void*)(Bt_));                   \
      async_b128(bL + 16, vaB + off + 16, (const void*)(Bt_));                   \
      /* async loads complete in order: <=4 left => current tile landed */       \
      asm volatile("s_wait_asynccnt 4" ::: "memory");                            \
    } else {                                                                     \
      asm volatile("s_wait_asynccnt 0" ::: "memory");                            \
    }                                                                            \
    __syncthreads(); /* tile `buf` visible to all waves */                       \
    const u16* Atb = At[buf];                                                    \
    const u16* Blb = Bl[buf];                                                    \
    Frag bfr[2];                                                                 \
    for (int j = 0; j < 2; ++j) {                                                \
      const u16* base = Blb + (no + j * 16 + ln) * 32;                           \
      bfr[j].q[0] = *(const uint4*)(base + lh * 8);                              \
      bfr[j].q[1] = *(const uint4*)(base + 16 + lh * 8);                         \
    }                                                                            \
    for (int i = 0; i < 4; ++i) {                                                \
      const u16* base = Atb + (mo + i * 16 + ln) * 32;                           \
      Frag af;                                                                   \
      af.q[0] = *(const uint4*)(base + lh * 8);                                  \
      af.q[1] = *(const uint4*)(base + 16 + lh * 8);                             \
      acc[i][0] = wmma_bf16(af.v, bfr[0].v, acc[i][0]);                          \
      acc[i][1] = wmma_bf16(af.v, bfr[1].v, acc[i][1]);                          \
    }                                                                            \
  }

// QKV projection: N = 3072, epilogue scatters to head-major Q, K and transposed V.
__launch_bounds__(256)
__global__ void gemm_qkv(const u16* __restrict__ A, const u16* __restrict__ Bt,
                         const float* __restrict__ bias,
                         u16* __restrict__ Qo, u16* __restrict__ Ko,
                         u16* __restrict__ Vto) {
  GEMM_MAINLOOP(A, Bt)
  for (int j = 0; j < 2; ++j) {
    const int ncol = nBlk + no + j * 16 + ln;        // 0..3071
    const float bv = bias[ncol];
    const int sel = ncol >> 10, f = ncol & 1023;
    const int h = f >> 6, dh = f & 63;
    for (int i = 0; i < 4; ++i)
      for (int r = 0; r < 8; ++r) {
        const int m = mBlk + mo + i * 16 + lh * 8 + r;  // 0..4095
        const int b = m >> 11, s = m & 2047;
        const u16 val = f32_to_bf16(acc[i][j][r] + bv);
        const size_t bh = (size_t)(b * 16 + h);
        if (sel == 0)      Qo[(bh * 2048 + s) * 64 + dh] = val;
        else if (sel == 1) Ko[(bh * 2048 + s) * 64 + dh] = val;
        else               Vto[(bh * 64 + dh) * 2048 + s] = val;
      }
  }
}

// Output projection: N = 1024, fp32 epilogue + bias to d_out.
__launch_bounds__(256)
__global__ void gemm_out(const u16* __restrict__ A, const u16* __restrict__ Bt,
                         const float* __restrict__ bias, float* __restrict__ out) {
  GEMM_MAINLOOP(A, Bt)
  for (int j = 0; j < 2; ++j) {
    const int ncol = nBlk + no + j * 16 + ln;
    const float bv = bias[ncol];
    for (int i = 0; i < 4; ++i)
      for (int r = 0; r < 8; ++r) {
        const int m = mBlk + mo + i * 16 + lh * 8 + r;
        out[(size_t)m * 1024 + ncol] = acc[i][j][r] + bv;
      }
  }
}

// ---------------------------------------------------------------- attention
// One wave per (b, h, 32-row q-tile). Flash-style online softmax; all 32 lanes
// own one q-row each in the softmax stage.
// Q,K: [B,H,S,64] bf16 row-major; Vt: [B,H,64,S] bf16; Ctx: [B*S, 1024] bf16.
__launch_bounds__(32)
__global__ void attn(const u16* __restrict__ Q, const u16* __restrict__ Kk,
                     const u16* __restrict__ Vt, const unsigned char* __restrict__ pad,
                     u16* __restrict__ Ctx) {
  __shared__ __align__(16) float sS[32 * 64];
  __shared__ __align__(16) u16   sP[32 * 64];
  __shared__ float sM[32], sL[32], sR[32];
  __shared__ __align__(16) unsigned char sPad[2048];

  const int tid = threadIdx.x;
  const int ln = tid & 15, lh = tid >> 4;
  const int qt = blockIdx.x & 63;
  const int h  = (blockIdx.x >> 6) & 15;
  const int b  = blockIdx.x >> 10;
  const int q0 = qt * 32;
  const size_t bh = (size_t)(b * 16 + h);
  const u16* Qp = Q  + bh * 2048 * 64;
  const u16* Kp = Kk + bh * 2048 * 64;
  const u16* Vp = Vt + bh * 64 * 2048;

  { // stage padding-mask row for this batch into LDS
    const uint4* src = (const uint4*)(pad + (size_t)b * 2048);
    uint4* dst = (uint4*)sPad;
    for (int i = 0; i < 4; ++i) dst[tid * 4 + i] = src[tid * 4 + i];
  }
  sM[tid] = -1e30f;
  sL[tid] = 0.f;
  __syncthreads();

  // Q fragments (A operand): 2 row tiles x 2 dh windows, reused for all k-blocks.
  Frag qa[2][2];
  for (int mt = 0; mt < 2; ++mt) {
    const u16* qrow = Qp + (size_t)(q0 + mt * 16 + ln) * 64;
    for (int w = 0; w < 2; ++w) {
      qa[mt][w].q[0] = *(const uint4*)(qrow + w * 32 + lh * 8);
      qa[mt][w].q[1] = *(const uint4*)(qrow + w * 32 + 16 + lh * 8);
    }
  }

  v8f o[2][4] = {};
  const int nkb = (q0 + 31) / 64 + 1;   // causal: only k-blocks with k <= q
  for (int kb = 0; kb < nkb; ++kb) {
    const int k0 = kb * 64;
    // ---- scores S = Q K^T / 8 : 4 column tiles of 16 keys
    for (int j = 0; j < 4; ++j) {
      const u16* krow = Kp + (size_t)(k0 + j * 16 + ln) * 64;
      Frag bw[2];
      for (int w = 0; w < 2; ++w) {
        bw[w].q[0] = *(const uint4*)(krow + w * 32 + lh * 8);
        bw[w].q[1] = *(const uint4*)(krow + w * 32 + 16 + lh * 8);
      }
      for (int mt = 0; mt < 2; ++mt) {
        v8f s = {};
        s = wmma_bf16(qa[mt][0].v, bw[0].v, s);
        s = wmma_bf16(qa[mt][1].v, bw[1].v, s);
        for (int r = 0; r < 8; ++r)
          sS[(mt * 16 + lh * 8 + r) * 64 + j * 16 + ln] = s[r] * 0.125f;
      }
    }
    __syncthreads();
    // ---- online softmax update: one lane per q-row
    {
      const int qg = q0 + tid;
      const float mOld = sM[tid];
      float mNew = mOld;
      for (int c = 0; c < 64; ++c) {
        const int k = k0 + c;
        if (k <= qg && !sPad[k]) mNew = fmaxf(mNew, sS[tid * 64 + c]);
      }
      const float sc = __expf(mOld - mNew);
      float ls = 0.f;
      for (int c = 0; c < 64; ++c) {
        const int k = k0 + c;
        float e = 0.f;
        if (k <= qg && !sPad[k]) e = __expf(sS[tid * 64 + c] - mNew);
        sP[tid * 64 + c] = f32_to_bf16(e);
        ls += e;
      }
      sL[tid] = sL[tid] * sc + ls;
      sM[tid] = mNew;
      sR[tid] = sc;
    }
    __syncthreads();
    // ---- rescale accumulators, O += P V
    float rs[2][8];
    for (int mt = 0; mt < 2; ++mt)
      for (int r = 0; r < 8; ++r) rs[mt][r] = sR[mt * 16 + lh * 8 + r];
    Frag pa[2][2];
    for (int mt = 0; mt < 2; ++mt) {
      const u16* prow = sP + (mt * 16 + ln) * 64;
      for (int w = 0; w < 2; ++w) {
        pa[mt][w].q[0] = *(const uint4*)(prow + w * 32 + lh * 8);
        pa[mt][w].q[1] = *(const uint4*)(prow + w * 32 + 16 + lh * 8);
      }
    }
    for (int j = 0; j < 4; ++j) {
      const u16* vrow = Vp + (size_t)(j * 16 + ln) * 2048 + k0;
      Frag bw[2];
      for (int w = 0; w < 2; ++w) {
        bw[w].q[0] = *(const uint4*)(vrow + w * 32 + lh * 8);
        bw[w].q[1] = *(const uint4*)(vrow + w * 32 + 16 + lh * 8);
      }
      for (int mt = 0; mt < 2; ++mt) {
        for (int r = 0; r < 8; ++r) o[mt][j][r] *= rs[mt][r];
        o[mt][j] = wmma_bf16(pa[mt][0].v, bw[0].v, o[mt][j]);
        o[mt][j] = wmma_bf16(pa[mt][1].v, bw[1].v, o[mt][j]);
      }
    }
    __syncthreads();
  }

  sR[tid] = 1.0f / sL[tid];
  __syncthreads();
  for (int mt = 0; mt < 2; ++mt) {
    float inv[8];
    for (int r = 0; r < 8; ++r) inv[r] = sR[mt * 16 + lh * 8 + r];
    for (int j = 0; j < 4; ++j)
      for (int r = 0; r < 8; ++r) {
        const int srow = q0 + mt * 16 + lh * 8 + r;
        const int col  = h * 64 + j * 16 + ln;
        Ctx[((size_t)(b * 2048 + srow)) * 1024 + col] = f32_to_bf16(o[mt][j][r] * inv[r]);
      }
  }
}

// ---------------------------------------------------------------- launch
extern "C" void kernel_launch(void* const* d_in, const int* in_sizes, int n_in,
                              void* d_out, int out_size, void* d_ws, size_t ws_size,
                              hipStream_t stream) {
  const float* query = (const float*)d_in[0];
  // d_in[1]=key, d_in[2]=value ignored by the reference (qkv proj of query only)
  const unsigned char* pad = (const unsigned char*)d_in[3]; // bool [B,S]
  const float* Wqkv = (const float*)d_in[4];
  const float* bqkv = (const float*)d_in[5];
  const float* Wout = (const float*)d_in[6];
  const float* bout = (const float*)d_in[7];
  float* out = (float*)d_out;

  u16* ws     = (u16*)d_ws;
  u16* Xbf    = ws;                         // 4096*1024
  u16* Wqkvb  = Xbf   + 4096 * 1024;        // 3072*1024
  u16* Woutb  = Wqkvb + 3072 * 1024;        // 1024*1024
  u16* Qb     = Woutb + 1024 * 1024;        // 4096*1024  [B,H,S,DH]
  u16* Kb     = Qb    + 4096 * 1024;        // 4096*1024  [B,H,S,DH]
  u16* Vtb    = Kb    + 4096 * 1024;        // 4096*1024  [B,H,DH,S]
  u16* Ctx    = Vtb   + 4096 * 1024;        // 4096*1024  [B*S, D]
  // total 24M u16 = 48 MB of d_ws

  cvt_f32_bf16<<<2048, 256, 0, stream>>>(query, Xbf,   4096 * 1024);
  cvt_f32_bf16<<<2048, 256, 0, stream>>>(Wqkv,  Wqkvb, 3072 * 1024);
  cvt_f32_bf16<<<1024, 256, 0, stream>>>(Wout,  Woutb, 1024 * 1024);

  gemm_qkv<<<dim3(24, 32), 256, 0, stream>>>(Xbf, Wqkvb, bqkv, Qb, Kb, Vtb);
  attn<<<2048, 32, 0, stream>>>(Qb, Kb, Vtb, pad, Ctx);
  gemm_out<<<dim3(8, 32), 256, 0, stream>>>(Ctx, Woutb, bout, out);
}
